// ShallowRegressionRNN_44487271252617
// MI455X (gfx1250) — matrix operand order repaired and checked by reading, not compiled, not run on hardware
//
#include <hip/hip_runtime.h>
#include <hip/hip_bf16.h>
#include <stdint.h>

// Problem constants (from reference)
#define BB   8192
#define TT   96
#define LL   24
#define HH   512

// Tiling
#define BT       64            // batch rows per block
#define XCOLS    (TT + LL - 1) // 119
#define XP       136           // x LDS stride (bf16 elems)
#define HP       (HH + 16)     // h LDS stride (bf16 elems), 1056B -> bank-friendly
#define WKP      40            // weight-slab LDS row stride (bf16), 80B = 64B data + 16B pad
#define SLABELEMS (HH * WKP)   // one K=32 x N=512 slab: 20480 elems = 40KB
#define NTHREADS 512           // 16 wave32s
#define NSLAB1   (TT / 32 + HH / 32)  // 19
#define NSLAB2   (2 * HH / 32)        // 32

// Prefer the Tensor Data Mover (TENSORcnt) for weight staging; fall back to
// per-lane async global->LDS (ASYNCcnt) if the builtin is unavailable.
#if defined(__AMDGCN__) && __has_builtin(__builtin_amdgcn_tensor_load_to_lds) && \
    __has_builtin(__builtin_amdgcn_s_wait_tensorcnt)
#define USE_TDM 1
#else
#define USE_TDM 0
#endif

typedef __attribute__((ext_vector_type(16))) __bf16 v16bf;
typedef __attribute__((ext_vector_type(8)))  float  v8f;
typedef __attribute__((ext_vector_type(4)))  unsigned int v4u;
typedef __attribute__((ext_vector_type(8)))  int    v8i;
typedef __attribute__((ext_vector_type(4)))  int    v4i;

__device__ __forceinline__ unsigned short f2bf(float f) {
  union { float f; unsigned u; } a; a.f = f;
  unsigned r = a.u + 0x7FFFu + ((a.u >> 16) & 1u);   // round-to-nearest-even
  return (unsigned short)(r >> 16);
}
__device__ __forceinline__ float bf2f(unsigned short u) {
  union { unsigned u; float f; } a; a.u = ((unsigned)u) << 16;
  return a.f;
}
__device__ __forceinline__ float tanh_fast(float x) {
  x = fminf(fmaxf(x, -15.f), 15.f);
  float e = __expf(2.f * x);                         // v_exp_f32
  return (e - 1.f) / (e + 1.f);
}

// ---- WMMA fragment loaders -------------------------------------------------
// A bf16 16x32 (contiguous layout): lane m = lane&15; VGPR r: K=2r,2r+1
// (+16 for lanes 16-31), matching the B slab fragment.
__device__ __forceinline__ v16bf load_a_aligned(const unsigned short* base, int stride,
                                                int row, int kbase, int half) {
  union { v16bf v; unsigned d[8]; } u;
  const unsigned short* p = base + row * stride + kbase + 16 * half;
#pragma unroll
  for (int r = 0; r < 8; ++r) {
    int k = 2 * r;
    u.d[r] = *(const unsigned*)(p + k);              // ds_load_b32
  }
  return u.v;
}
// Element-wise variant (x window start t may be odd -> only 2B aligned)
__device__ __forceinline__ v16bf load_a_u16(const unsigned short* base, int stride,
                                            int row, int kbase, int half) {
  union { v16bf v; unsigned short s[16]; } u;
  const unsigned short* p = base + row * stride + kbase + 16 * half;
#pragma unroll
  for (int r = 0; r < 8; ++r) {
    int k = 2 * r;
    u.s[2 * r]     = p[k];
    u.s[2 * r + 1] = p[k + 1];
  }
  return u.v;
}
// B bf16 32x16 fragment from the LDS weight slab (row j holds K=0..31 of W^T
// column j). lanes 0-15: K=0..15, lanes 16-31: K=16..31.
__device__ __forceinline__ v16bf load_wslab(const unsigned short* wb, int j, int half) {
  union { v16bf v; uint4 q[2]; } u;
  const unsigned short* p = wb + j * WKP + 16 * half;
  u.q[0] = *(const uint4*)(p);                       // ds_load_b128
  u.q[1] = *(const uint4*)(p + 8);                   // ds_load_b128
  return u.v;
}

// ---- Weight-slab staging: global (bf16, row-major [j, k]) -> LDS slab ------
// Slab = rows j=0..511, K columns [kbase, kbase+32): 512 rows x 64B, written
// into LDS with an 80B pitch (16B pad per row for bank-conflict-free b128).
#if USE_TDM
// One Tensor Data Mover descriptor moves the whole 32KB slab; the hardware
// applies the 16B/row LDS padding (pad_interval=16 DWORDs, pad_amount=4 DWORDs).
__device__ __forceinline__ void stage_slab_tdm(const unsigned short* __restrict__ Wg,
                                               int Kw, int kbase, unsigned lds_byteoff) {
  unsigned long long ga = (unsigned long long)(const void*)(Wg + kbase);
  int kd0 = Kw >> 2;                       // row length / stride in 8B units
  v4u g0;
  g0[0] = 1u;                              // count=1 valid user descriptor
  g0[1] = lds_byteoff;                     // lds_addr (bytes)
  g0[2] = (unsigned)(ga & 0xFFFFFFFFu);    // global_addr[31:0]
  g0[3] = (unsigned)((ga >> 32) & 0x01FFFFFFu) | (2u << 30);  // addr[56:32], type=2
  v8i g1;
  g1[0] = (3 << 16)      // data_size = 8B
        | (1 << 20)      // pad_enable
        | (3 << 22)      // pad_interval: 16 DWORDs (64B) between pads
        | (3 << 25);     // pad_amount: 4 DWORDs (16B)
  g1[1] = (kd0 & 0xFFFF) << 16;            // tensor_dim0[15:0] (atomic_addr=0)
  g1[2] = (kd0 >> 16) | (HH << 16);        // tensor_dim0[31:16] | tensor_dim1[15:0]=512
  g1[3] = 8 << 16;                         // tile_dim0 = 8 (x8B = 64B row)
  g1[4] = HH;                              // tile_dim1 = 512 rows, tile_dim2 = 0
  g1[5] = kd0;                             // tensor_dim0_stride[31:0]
  g1[6] = 0;
  g1[7] = 0;
  v4i z4 = {0, 0, 0, 0};
  v8i z8 = {0, 0, 0, 0, 0, 0, 0, 0};
  // 6-arg toolchain form: (g0, g1, g2, g3, extra, cpol); tracked on TENSORcnt
  __builtin_amdgcn_tensor_load_to_lds(g0, g1, z4, z4, z8, 0);
}
__device__ __forceinline__ void stage_slab(const unsigned short* __restrict__ Wg, int Kw,
                                           int kbase, unsigned lds_byteoff, int tid, int wave) {
  if (wave == 0) stage_slab_tdm(Wg, Kw, kbase, lds_byteoff);
}
__device__ __forceinline__ void wait_stage(int wave) {
  if (wave == 0) __builtin_amdgcn_s_wait_tensorcnt(0);
}
#else
__device__ __forceinline__ void stage_slab(const unsigned short* __restrict__ Wg, int Kw,
                                           int kbase, unsigned lds_byteoff, int tid, int wave) {
#pragma unroll
  for (int i = 0; i < 4; ++i) {
    int idx = tid + i * NTHREADS;                    // 0..2047 chunks of 16B
    int j = idx >> 2, c = idx & 3;
    const unsigned short* g = Wg + (size_t)j * Kw + kbase + c * 8;
    unsigned loff = lds_byteoff + (unsigned)(j * WKP + c * 8) * 2u;
    asm volatile("global_load_async_to_lds_b128 %0, %1, off"
                 :: "v"(loff), "v"(g) : "memory");
  }
}
__device__ __forceinline__ void wait_stage(int wave) {
  asm volatile("s_wait_asynccnt 0x0" ::: "memory");
}
#endif

// One K=32 slab of the GEMM: 8 WMMAs on this wave's 8 column tiles.
// B-fragments are prefetched 4 at a time so one DScnt wait covers 4 WMMAs.
template <bool ALIGNED>
__device__ __forceinline__ void compute_slab(v8f acc[8],
                                             const unsigned short* albase, int astride,
                                             int row, int akbase,
                                             const unsigned short* wb,
                                             int jg, int nn, int half) {
  v16bf a = ALIGNED ? load_a_aligned(albase, astride, row, akbase, half)
                    : load_a_u16(albase, astride, row, akbase, half);
#pragma unroll
  for (int g = 0; g < 2; ++g) {
    v16bf b[4];
#pragma unroll
    for (int i = 0; i < 4; ++i)
      b[i] = load_wslab(wb, jg + (g * 4 + i) * 16 + nn, half);
#pragma unroll
    for (int i = 0; i < 4; ++i)
      acc[g * 4 + i] = __builtin_amdgcn_wmma_f32_16x16x32_bf16(
          false, a, false, b[i], (short)0, acc[g * 4 + i], false, false);
  }
}

__global__ void cvt_bf16_kernel(const float* __restrict__ src,
                                unsigned short* __restrict__ dst, int n) {
  int i = blockIdx.x * blockDim.x + threadIdx.x;
  if (i < n) dst[i] = f2bf(src[i]);
}

__global__ void bias_kernel(const float* __restrict__ b_ih1, const float* __restrict__ b_hh1,
                            const float* __restrict__ b_ih2, const float* __restrict__ b_hh2,
                            const float* __restrict__ b_fc,  float* __restrict__ dst) {
  int i = blockIdx.x * blockDim.x + threadIdx.x;
  if (i < HH)            dst[i] = b_ih1[i] + b_hh1[i];
  else if (i < 2 * HH)   dst[i] = b_ih2[i - HH] + b_hh2[i - HH];
  else if (i == 2 * HH)  dst[i] = b_fc[0];
}

__global__ __launch_bounds__(NTHREADS)
void rnn_kernel(const float* __restrict__ x, const float* __restrict__ h1in,
                const float* __restrict__ h2in,
                const unsigned short* __restrict__ wsbf,
                const float* __restrict__ biases,
                float* __restrict__ out) {
  extern __shared__ unsigned short smem[];
  unsigned short* xlds  = smem;                         // BT * XP
  unsigned short* h1lds = xlds + BT * XP;               // BT * HP
  unsigned short* h2lds = h1lds + BT * HP;              // BT * HP
  unsigned short* wbuf  = h2lds + BT * HP;              // 2 * SLABELEMS
  const unsigned wbuf_byteoff = (unsigned)((BT * XP + 2 * BT * HP) * 2);

  const unsigned short* W1X = wsbf;                     // [512, 96]
  const unsigned short* W1H = W1X + HH * TT;            // [512,512]
  const unsigned short* W2I = W1H + HH * HH;            // [512,512]
  const unsigned short* W2H = W2I + HH * HH;            // [512,512]
  const unsigned short* WFC = W2H + HH * HH;            // [512]
  const float* bias1 = biases;
  const float* bias2 = biases + HH;

  const int tid  = threadIdx.x;
  const int lane = tid & 31;
  const int wave = tid >> 5;
  const int b0   = blockIdx.x * BT;

  const int mm   = lane & 15;          // A-row within tile
  const int nn   = lane & 15;          // B/D-col within tile
  const int half = lane >> 4;
  const int rb   = (wave & 3) * 16;    // row-tile base (0..48)
  const int jg   = (wave >> 2) * 128;  // first output column of this wave's 8 tiles

  // Stage x window and initial hidden states as bf16 in LDS
  for (int i = tid; i < BT * XCOLS; i += NTHREADS) {
    int rr = i / XCOLS, cc = i - rr * XCOLS;
    xlds[rr * XP + cc] = f2bf(x[(size_t)(b0 + rr) * XCOLS + cc]);
  }
  for (int i = tid; i < BT * HH; i += NTHREADS) {
    int rr = i >> 9, cc = i & (HH - 1);
    h1lds[rr * HP + cc] = f2bf(h1in[(size_t)(b0 + rr) * HH + cc]);
    h2lds[rr * HP + cc] = f2bf(h2in[(size_t)(b0 + rr) * HH + cc]);
  }
  __syncthreads();

  const float bfcv = biases[2 * HH];

#pragma unroll 1
  for (int t = 0; t < LL; ++t) {
    v8f acc[8];

    // ============ layer 1: h1 = tanh(xt@Wih1^T + h1@Whh1^T + b) ============
#pragma unroll
    for (int ci = 0; ci < 8; ++ci) {
      float bv = bias1[jg + ci * 16 + nn];
#pragma unroll
      for (int e = 0; e < 8; ++e) acc[ci][e] = bv;
    }
    // prologue: slab 0 into buffer 0
    stage_slab(W1X, TT, 0, wbuf_byteoff, tid, wave);
    wait_stage(wave);
    __syncthreads();
#pragma unroll 1
    for (int s = 0; s < NSLAB1; ++s) {
      int nxt = s + 1;
      if (nxt < NSLAB1) {        // DMA next slab while computing current one
        unsigned dst = wbuf_byteoff + (unsigned)(nxt & 1) * (SLABELEMS * 2);
        if (nxt < TT / 32) stage_slab(W1X, TT, nxt * 32, dst, tid, wave);
        else               stage_slab(W1H, HH, (nxt - TT / 32) * 32, dst, tid, wave);
      }
      const unsigned short* wb = wbuf + (s & 1) * SLABELEMS;
      if (s < TT / 32)
        compute_slab<false>(acc, xlds, XP, rb + mm, t + s * 32, wb, jg, nn, half);
      else
        compute_slab<true>(acc, h1lds, HP, rb + mm, (s - TT / 32) * 32, wb, jg, nn, half);
      if (nxt < NSLAB1) { wait_stage(wave); __syncthreads(); }
    }
    __syncthreads();   // all waves done reading h1(t-1)
#pragma unroll
    for (int ci = 0; ci < 8; ++ci) {
      int j = jg + ci * 16 + nn;
#pragma unroll
      for (int e = 0; e < 8; ++e)   // D layout: VGPR e -> row e (lanes 0-15), e+8 (16-31)
        h1lds[(rb + e + 8 * half) * HP + j] = f2bf(tanh_fast(acc[ci][e]));
    }
    __syncthreads();

    // ============ layer 2: h2 = tanh(h1@Wih2^T + h2@Whh2^T + b) ============
#pragma unroll
    for (int ci = 0; ci < 8; ++ci) {
      float bv = bias2[jg + ci * 16 + nn];
#pragma unroll
      for (int e = 0; e < 8; ++e) acc[ci][e] = bv;
    }
    stage_slab(W2I, HH, 0, wbuf_byteoff, tid, wave);
    wait_stage(wave);
    __syncthreads();
#pragma unroll 1
    for (int s = 0; s < NSLAB2; ++s) {
      int nxt = s + 1;
      if (nxt < NSLAB2) {
        unsigned dst = wbuf_byteoff + (unsigned)(nxt & 1) * (SLABELEMS * 2);
        if (nxt < HH / 32) stage_slab(W2I, HH, nxt * 32, dst, tid, wave);
        else               stage_slab(W2H, HH, (nxt - HH / 32) * 32, dst, tid, wave);
      }
      const unsigned short* wb = wbuf + (s & 1) * SLABELEMS;
      if (s < HH / 32)
        compute_slab<true>(acc, h1lds, HP, rb + mm, s * 32, wb, jg, nn, half);
      else
        compute_slab<true>(acc, h2lds, HP, rb + mm, (s - HH / 32) * 32, wb, jg, nn, half);
      if (nxt < NSLAB2) { wait_stage(wave); __syncthreads(); }
    }
    __syncthreads();   // all waves done reading h2(t-1)
#pragma unroll
    for (int ci = 0; ci < 8; ++ci) {
      int j = jg + ci * 16 + nn;
#pragma unroll
      for (int e = 0; e < 8; ++e)
        h2lds[(rb + e + 8 * half) * HP + j] = f2bf(tanh_fast(acc[ci][e]));
    }
    __syncthreads();

    // ============ fc head: y = h2 @ Wfc^T + b_fc ===========================
    {
      int row = tid >> 3, part = tid & 7;              // 8 lanes per batch row
      const unsigned short* hp = h2lds + row * HP + part * 64;
      float s = 0.f;
#pragma unroll 8
      for (int j = 0; j < 64; ++j) s += bf2f(hp[j]) * bf2f(WFC[part * 64 + j]);
      s += __shfl_xor(s, 1, 32);
      s += __shfl_xor(s, 2, 32);
      s += __shfl_xor(s, 4, 32);
      if (part == 0) out[(size_t)(b0 + row) * LL + t] = s + bfcv;
    }
    __syncthreads();
  }

  // Final hidden states back to global as fp32
  float* h1out = out + (size_t)BB * LL;
  float* h2out = h1out + (size_t)BB * HH;
  for (int i = tid; i < BT * HH; i += NTHREADS) {
    int rr = i >> 9, cc = i & (HH - 1);
    h1out[(size_t)(b0 + rr) * HH + cc] = bf2f(h1lds[rr * HP + cc]);
    h2out[(size_t)(b0 + rr) * HH + cc] = bf2f(h2lds[rr * HP + cc]);
  }
}

extern "C" void kernel_launch(void* const* d_in, const int* in_sizes, int n_in,
                              void* d_out, int out_size, void* d_ws, size_t ws_size,
                              hipStream_t stream) {
  const float* x    = (const float*)d_in[0];
  const float* h1in = (const float*)d_in[1];
  const float* h2in = (const float*)d_in[2];
  const float* Wih1 = (const float*)d_in[3];
  const float* Whh1 = (const float*)d_in[4];
  const float* bih1 = (const float*)d_in[5];
  const float* bhh1 = (const float*)d_in[6];
  const float* Wih2 = (const float*)d_in[7];
  const float* Whh2 = (const float*)d_in[8];
  const float* bih2 = (const float*)d_in[9];
  const float* bhh2 = (const float*)d_in[10];
  const float* Wfc  = (const float*)d_in[11];
  const float* bfc  = (const float*)d_in[12];

  // Workspace layout: bf16 weights, then fp32 fused biases
  unsigned short* wsbf = (unsigned short*)d_ws;
  const size_t nbf = (size_t)HH * TT + 3 * (size_t)HH * HH + HH;   // 836,096 bf16
  float* biases = (float*)((char*)d_ws + ((nbf * 2 + 255) & ~(size_t)255));

  unsigned short* W1X = wsbf;
  unsigned short* W1H = W1X + HH * TT;
  unsigned short* W2I = W1H + HH * HH;
  unsigned short* W2H = W2I + HH * HH;
  unsigned short* WFC = W2H + HH * HH;

  cvt_bf16_kernel<<<(HH * TT + 255) / 256, 256, 0, stream>>>(Wih1, W1X, HH * TT);
  cvt_bf16_kernel<<<(HH * HH + 255) / 256, 256, 0, stream>>>(Whh1, W1H, HH * HH);
  cvt_bf16_kernel<<<(HH * HH + 255) / 256, 256, 0, stream>>>(Wih2, W2I, HH * HH);
  cvt_bf16_kernel<<<(HH * HH + 255) / 256, 256, 0, stream>>>(Whh2, W2H, HH * HH);
  cvt_bf16_kernel<<<(HH + 255) / 256, 256, 0, stream>>>(Wfc, WFC, HH);
  bias_kernel<<<(2 * HH + 1 + 255) / 256, 256, 0, stream>>>(bih1, bhh1, bih2, bhh2, bfc, biases);

  // LDS: x window (17KB) + h1/h2 (2x66KB) + double-buffered weight slab (80KB)
  // = ~229KB of the WGP's 320KB; dynamic to bypass the 64KB static cap.
  size_t smem = (size_t)(BT * XP + 2 * BT * HP + 2 * SLABELEMS) * sizeof(unsigned short);
  rnn_kernel<<<BB / BT, NTHREADS, smem, stream>>>(x, h1in, h2in, wsbf, biases,
                                                  (float*)d_out);
}
